// LateralInhibitionLayer_78125455114739
// MI455X (gfx1250) — 8-bit matrix operand order repaired and checked by reading, compiled, not run on hardware
//
#include <hip/hip_runtime.h>
#include <stdint.h>

#define NN   4096
#define BT   8192

typedef int   v16i __attribute__((ext_vector_type(16)));
typedef float v8f  __attribute__((ext_vector_type(8)));

__device__ __forceinline__ unsigned f32_to_e4m3(float x) {
    unsigned u = __float_as_uint(x);
    unsigned s = (u >> 24) & 0x80u;
    unsigned a = u & 0x7fffffffu;
    if (a >= 0x7f800000u) return s | 0x7fu;
    if (a >= 0x43e80000u) return s | 0x7eu;
    int e = (int)(a >> 23) - 127;
    if (e >= -6) {
        unsigned mant = a & 0x7fffffu;
        unsigned v = ((unsigned)(e + 7) << 3) | (mant >> 20);
        unsigned rem = mant & 0xfffffu;
        if (rem > 0x80000u || (rem == 0x80000u && (v & 1u))) v++;
        return s | (v > 0x7eu ? 0x7eu : v);
    }
    float k = __uint_as_float(a) * 512.0f;
    unsigned q = (unsigned)(k + 0.5f);
    if (q > 8u) q = 8u;
    return s | q;
}

__global__ __launch_bounds__(256) void wcvt_kernel(const float* __restrict__ W,
                                                   unsigned char* __restrict__ W8) {
    size_t i = ((size_t)blockIdx.x * 256u + threadIdx.x) * 4u;
    float4 w = *(const float4*)(W + i);
    unsigned p = f32_to_e4m3(w.x) | (f32_to_e4m3(w.y) << 8) |
                 (f32_to_e4m3(w.z) << 16) | (f32_to_e4m3(w.w) << 24);
    *(unsigned*)(W8 + i) = p;
}

__global__ __launch_bounds__(256) void winners_kernel(const float* __restrict__ M,
                                                      unsigned char* __restrict__ Wn) {
    __shared__ float red[256];
    const unsigned tid  = threadIdx.x;
    const size_t   base = (size_t)blockIdx.x * NN + tid * 16u;
    float4 v[4];
    float mx = -3.402823466e38f;
#pragma unroll
    for (int j = 0; j < 4; ++j) {
        v[j] = *(const float4*)(M + base + j * 4);
        mx = fmaxf(mx, fmaxf(fmaxf(v[j].x, v[j].y), fmaxf(v[j].z, v[j].w)));
    }
    red[tid] = mx;
    __syncthreads();
#pragma unroll
    for (int s = 128; s > 0; s >>= 1) {
        if (tid < (unsigned)s) red[tid] = fmaxf(red[tid], red[tid + s]);
        __syncthreads();
    }
    const float rm = red[0];
    uint4 o;
    unsigned* po = (unsigned*)&o;
#pragma unroll
    for (int j = 0; j < 4; ++j) {
        po[j] = (v[j].x == rm ? 0x38u : 0u)        |
                ((v[j].y == rm ? 0x38u : 0u) << 8)  |
                ((v[j].z == rm ? 0x38u : 0u) << 16) |
                ((v[j].w == rm ? 0x38u : 0u) << 24);
    }
    *(uint4*)(Wn + base) = o;
}

__device__ __forceinline__ void stage_tile(unsigned lds_base, unsigned tid,
                                           unsigned rowBase, unsigned k0,
                                           unsigned long long gbase) {
#pragma unroll
    for (int it = 0; it < 4; ++it) {
        unsigned idx  = (unsigned)it * 4096u + tid * 16u;
        unsigned r    = idx >> 7, c = idx & 127u;
        unsigned voff = (rowBase + r) * (unsigned)NN + k0 + c;
        asm volatile("global_load_async_to_lds_b128 %0, %1, %2"
                     :: "v"(lds_base + idx), "v"(voff), "s"(gbase) : "memory");
    }
}

__global__ __launch_bounds__(256) void gemm_kernel(const float* __restrict__ membrane,
                                                   const unsigned char* __restrict__ A8,
                                                   const unsigned char* __restrict__ B8,
                                                   float* __restrict__ out) {
    __shared__ __align__(16) unsigned char ldsA[2][128 * 128];
    __shared__ __align__(16) unsigned char ldsB[2][128 * 128];

    const unsigned tid  = threadIdx.x;
    const unsigned lane = tid & 31u;
    const unsigned wave = tid >> 5;
    const unsigned wm   = wave & 3u;
    const unsigned wn   = wave >> 2;
    const unsigned m    = lane & 15u;
    const unsigned hi   = lane >> 4;

    const unsigned rowBase = blockIdx.y * 128u;
    const unsigned colBase = blockIdx.x * 128u;

    const unsigned ldsA_off = (unsigned)(size_t)&ldsA[0][0];
    const unsigned ldsB_off = (unsigned)(size_t)&ldsB[0][0];
    const unsigned long long aBase = (unsigned long long)A8;
    const unsigned long long bBase = (unsigned long long)B8;

    v8f acc[2][4] = {};

    stage_tile(ldsA_off, tid, rowBase, 0u, aBase);
    stage_tile(ldsB_off, tid, colBase, 0u, bBase);

    const int KSTEPS = NN / 128;
    for (int k = 0; k < KSTEPS; ++k) {
        const unsigned cur = (unsigned)k & 1u;
        if (k + 1 < KSTEPS) {
            const unsigned nxt = cur ^ 1u;
            const unsigned k0n = (unsigned)(k + 1) * 128u;
            stage_tile(ldsA_off + nxt * 16384u, tid, rowBase, k0n, aBase);
            stage_tile(ldsB_off + nxt * 16384u, tid, colBase, k0n, bBase);
            asm volatile("s_wait_asynccnt 0x8" ::: "memory");
        } else {
            asm volatile("s_wait_asynccnt 0x0" ::: "memory");
        }
        __syncthreads();

        const unsigned aBuf = ldsA_off ? 0u : 0u;
        const unsigned char* __restrict__ pA = &ldsA[cur][0];
        const unsigned char* __restrict__ pB = &ldsB[cur][0];
        (void)aBuf;

        v16i a[2], b[4];
#pragma unroll
        for (int mi = 0; mi < 2; ++mi) {
            const unsigned row = wm * 32u + (unsigned)mi * 16u + m;
#pragma unroll
            for (int v = 0; v < 16; ++v) {
                unsigned kb = (((unsigned)v >> 2) & 3u) * 32u + hi * 16u + ((unsigned)v & 3u) * 4u;
                a[mi][v] = *(const int*)&pA[row * 128u + kb];
            }
        }
#pragma unroll
        for (int ni = 0; ni < 4; ++ni) {
            const unsigned col = wn * 64u + (unsigned)ni * 16u + m;
#pragma unroll
            for (int v = 0; v < 16; ++v) {
                unsigned kb = (((unsigned)v >> 2) & 3u) * 32u + hi * 16u + ((unsigned)v & 3u) * 4u;
                b[ni][v] = *(const int*)&pB[col * 128u + kb];
            }
        }

#pragma unroll
        for (int mi = 0; mi < 2; ++mi)
#pragma unroll
            for (int ni = 0; ni < 4; ++ni)
                acc[mi][ni] = __builtin_amdgcn_wmma_f32_16x16x128_fp8_fp8(
                    a[mi], b[ni], (short)0, acc[mi][ni], false, false);

        __syncthreads();
    }

#pragma unroll
    for (int mi = 0; mi < 2; ++mi) {
#pragma unroll
        for (int ni = 0; ni < 4; ++ni) {
#pragma unroll
            for (int v = 0; v < 8; ++v) {
                unsigned gr = rowBase + wm * 32u + (unsigned)mi * 16u + (unsigned)v + hi * 8u;
                unsigned gc = colBase + wn * 64u + (unsigned)ni * 16u + m;
                size_t o = (size_t)gr * NN + gc;
                out[o] = membrane[o] - acc[mi][ni][v];
            }
        }
    }
}

extern "C" void kernel_launch(void* const* d_in, const int* in_sizes, int n_in,
                              void* d_out, int out_size, void* d_ws, size_t ws_size,
                              hipStream_t stream) {
    const float* membrane = (const float*)d_in[0];
    const float* W        = (const float*)d_in[1];
    float*       out      = (float*)d_out;

    unsigned char* winners8 = (unsigned char*)d_ws;
    unsigned char* W8       = (unsigned char*)d_ws + (size_t)BT * NN;

    wcvt_kernel<<<(NN * NN) / (256 * 4), 256, 0, stream>>>(W, W8);
    winners_kernel<<<BT, 256, 0, stream>>>(membrane, winners8);
    gemm_kernel<<<dim3(NN / 128, BT / 128), 256, 0, stream>>>(membrane, winners8, W8, out);
}
